// CausalSelfAttention_46291157516839
// MI455X (gfx1250) — matrix operand order repaired and checked by reading, not compiled, not run on hardware
//
#include <hip/hip_runtime.h>
#include <hip/hip_bf16.h>
#include <stdint.h>

// ---------------------------------------------------------------------------
// CDNA5 (gfx1250) causal self-attention: bf16 WMMA pipeline, wave32.
//   K1: QKV = x @ w_qkv   (fp32 -> bf16)  double-buffered WMMA GEMM
//   K2: attention per (b,h) (bf16)        async global->LDS + WMMA + softmax
//   K3: out = attn @ w_proj (bf16 -> f32) double-buffered WMMA GEMM
// ---------------------------------------------------------------------------

typedef __attribute__((ext_vector_type(16))) __bf16 v16bf;
typedef __attribute__((ext_vector_type(8)))  float  v8f;

__device__ __forceinline__ __bf16 cvt_bf16(float f)          { return (__bf16)f; }
__device__ __forceinline__ __bf16 cvt_bf16(unsigned short u) { return __builtin_bit_cast(__bf16, u); }
__device__ __forceinline__ unsigned short bf_bits(__bf16 v)  { return __builtin_bit_cast(unsigned short, v); }

// Async 16-byte copy global -> LDS (GV mode, ASYNCcnt-tracked).
__device__ __forceinline__ void async_copy_b128(uint32_t lds_byte_off, const void* gptr) {
  uint64_t ga = (uint64_t)(uintptr_t)gptr;
  asm volatile("global_load_async_to_lds_b128 %0, %1, off"
               :: "v"(lds_byte_off), "v"(ga) : "memory");
}
__device__ __forceinline__ void async_wait_all() {
  asm volatile("s_wait_asynccnt 0" ::: "memory");
}

// A-matrix fragment (16x32 bf16, MxK) from LDS, row-major with stride ld.
// Contiguous k (this code): lanes 0-15 -> M=lane,    K = 0..15 (+k0)
//      lanes16-31 -> M=lane-16, K = 16..31
__device__ __forceinline__ v16bf lds_frag_a(const __bf16* base, int ld, int m0, int k0) {
  const int lane = threadIdx.x & 31;
  const __bf16* p = base + (size_t)(m0 + (lane & 15)) * ld + k0 + ((lane >> 4) << 4);
  v16bf f;
#pragma unroll
  for (int i = 0; i < 16; ++i) f[i] = p[i];
  return f;
}

// B-matrix fragment (32x16 bf16, KxN) from LDS stored N-major (Bt[n][k], stride ld).
// Contiguous k (as A): lanes 0-15 -> N=lane, K=0..15; lanes 16-31 -> K=16..31.
__device__ __forceinline__ v16bf lds_frag_b(const __bf16* bt, int ld, int n0, int k0) {
  const int lane = threadIdx.x & 31;
  const __bf16* p = bt + (size_t)(n0 + (lane & 15)) * ld + k0 + ((lane >> 4) << 4);
  v16bf f;
#pragma unroll
  for (int i = 0; i < 16; ++i) f[i] = p[i];
  return f;
}

// ---------------------------------------------------------------------------
// Tiled GEMM: C[M,N] = A[M,K] * B[K,N].  Block tile 128x128, BK=32,
// register-staged double-buffered LDS so the next tile's HBM fetch overlaps
// the 8 WMMAs of the current tile.  8 waves, each 32x64 (2x4 WMMA tiles).
// ---------------------------------------------------------------------------
template <typename TA, typename TB, bool OUT_BF16>
__global__ __launch_bounds__(256) void gemm_bf16_wmma(
    const TA* __restrict__ A, const TB* __restrict__ B, void* __restrict__ Cout,
    int N, int K) {
  const int lda = K, ldb = N, ldc = N;
  __shared__ __bf16 sA [2][128 * 40];   // A tile 128x32, pad to 40
  __shared__ __bf16 sBt[2][128 * 40];   // B tile 32x128 stored N-major

  const int tid  = threadIdx.x;
  const int wave = tid >> 5, lane = tid & 31;
  const int mBlk = blockIdx.x * 128, nBlk = blockIdx.y * 128;
  const int wm = (wave >> 1) * 32;   // 0,32,64,96
  const int wn = (wave & 1) * 64;    // 0,64

  v8f acc[2][4];
#pragma unroll
  for (int i = 0; i < 2; ++i)
#pragma unroll
    for (int j = 0; j < 4; ++j)
#pragma unroll
      for (int r = 0; r < 8; ++r) acc[i][j][r] = 0.0f;

  TA va[16];   // staged A elements for the in-flight tile
  TB vb[16];   // staged B elements

  auto fetch = [&](int k0) {
#pragma unroll
    for (int i = 0; i < 16; ++i) {
      int idx = i * 256 + tid;                 // 0..4095
      int ra = idx >> 5, ca = idx & 31;        // A: 128 x 32
      va[i] = A[(size_t)(mBlk + ra) * lda + k0 + ca];
      int rb = idx >> 7, cb = idx & 127;       // B: 32 x 128
      vb[i] = B[(size_t)(k0 + rb) * ldb + nBlk + cb];
    }
  };
  auto stage = [&](int buf) {
#pragma unroll
    for (int i = 0; i < 16; ++i) {
      int idx = i * 256 + tid;
      int ra = idx >> 5, ca = idx & 31;
      sA[buf][ra * 40 + ca] = cvt_bf16(va[i]);
      int rb = idx >> 7, cb = idx & 127;
      sBt[buf][cb * 40 + rb] = cvt_bf16(vb[i]);
    }
  };

  fetch(0);
  const int KT = K / 32;
  for (int kt = 0; kt < KT; ++kt) {
    const int cur = kt & 1;
    stage(cur);
    __syncthreads();
    if (kt + 1 < KT) fetch((kt + 1) * 32);   // HBM fetch overlaps WMMAs below

    v16bf af[2], bq[4];
#pragma unroll
    for (int i = 0; i < 2; ++i) af[i] = lds_frag_a(sA[cur], 40, wm + 16 * i, 0);
#pragma unroll
    for (int j = 0; j < 4; ++j) bq[j] = lds_frag_b(sBt[cur], 40, wn + 16 * j, 0);
#pragma unroll
    for (int i = 0; i < 2; ++i)
#pragma unroll
      for (int j = 0; j < 4; ++j)
        acc[i][j] = __builtin_amdgcn_wmma_f32_16x16x32_bf16(
            false, af[i], false, bq[j], (short)0, acc[i][j], false, false);
    // no second barrier needed: next iteration stages into the other buffer,
    // and nobody can be 2 iterations ahead without passing the next barrier.
  }

  // C/D layout: VGPR r, lanes 0-15 -> M=r, N=lane; lanes 16-31 -> M=8+r.
  const int lc = lane & 15, lr = (lane >> 4) << 3;
#pragma unroll
  for (int i = 0; i < 2; ++i)
#pragma unroll
    for (int j = 0; j < 4; ++j)
#pragma unroll
      for (int r = 0; r < 8; ++r) {
        size_t row = (size_t)(mBlk + wm + 16 * i + lr + r);
        size_t col = (size_t)(nBlk + wn + 16 * j + lc);
        float v = acc[i][j][r];
        if constexpr (OUT_BF16)
          ((unsigned short*)Cout)[row * ldc + col] = bf_bits((__bf16)v);
        else
          ((float*)Cout)[row * ldc + col] = v;
      }
}

// ---------------------------------------------------------------------------
// Attention per (b,h): S = Q K^T / 8, causal mask, softmax, O = P V.
// Q/K tiles are pulled into LDS with GLOBAL_LOAD_ASYNC_TO_LDS_B128 (no VGPR
// staging, ASYNCcnt-tracked); V goes through VGPRs because it is transposed.
// ---------------------------------------------------------------------------
__global__ __launch_bounds__(256) void attn_wmma(const unsigned short* __restrict__ qkv_u,
                                                 unsigned short* __restrict__ attnout) {
  const int T = 128, C3 = 2304;
  const int b = blockIdx.x, h = blockIdx.y;
  const int tid  = threadIdx.x;
  const int wave = tid >> 5, lane = tid & 31;
  const int lc = lane & 15, lr = (lane >> 4) << 3;
  const int m0 = wave * 16;

  // Phase A: sQ(128x72) + sK(128x72).  Phase B (aliased): sP(128x136) + sVt(64x136).
  __shared__ __align__(16) char smem[52224];
  __bf16* sQ  = (__bf16*)smem;
  __bf16* sK  = (__bf16*)(smem + 18432);
  __bf16* sP  = (__bf16*)smem;
  __bf16* sVt = (__bf16*)(smem + 34816);

  const __bf16* baseq = (const __bf16*)qkv_u + (size_t)b * T * C3 + h * 64;
  const uint32_t sQoff = (uint32_t)(uintptr_t)(void*)sQ;   // low 32 bits = LDS byte addr
  const uint32_t sKoff = (uint32_t)(uintptr_t)(void*)sK;

  // ---- async copy Q and K tiles into LDS (rows are 128B = 8 x b128) ----
#pragma unroll
  for (int i = 0; i < 4; ++i) {
    int idx = i * 256 + tid;            // 0..1023
    int t = idx >> 3, c = idx & 7;      // row, 16-byte chunk
    async_copy_b128(sQoff + t * 144 + c * 16, baseq + (size_t)t * C3 + c * 8);
    async_copy_b128(sKoff + t * 144 + c * 16, baseq + (size_t)t * C3 + 768 + c * 8);
  }
  async_wait_all();
  __syncthreads();

  // ---- S = Q K^T : 8 column tiles, 2 K-steps over D=64 ----
  v8f sacc[8];
#pragma unroll
  for (int j = 0; j < 8; ++j)
#pragma unroll
    for (int r = 0; r < 8; ++r) sacc[j][r] = 0.0f;

  for (int k0 = 0; k0 < 64; k0 += 32) {
    v16bf aq = lds_frag_a(sQ, 72, m0, k0);
#pragma unroll
    for (int j = 0; j < 8; ++j) {
      v16bf bk = lds_frag_b(sK, 72, 16 * j, k0);
      sacc[j] = __builtin_amdgcn_wmma_f32_16x16x32_bf16(
          false, aq, false, bk, (short)0, sacc[j], false, false);
    }
  }

  // ---- causal mask + softmax (rows distributed per C-layout) ----
  float mx[8], sm[8];
#pragma unroll
  for (int r = 0; r < 8; ++r) mx[r] = -3.0e38f;
#pragma unroll
  for (int j = 0; j < 8; ++j)
#pragma unroll
    for (int r = 0; r < 8; ++r) {
      float sv = sacc[j][r] * 0.125f;              // 1/sqrt(64)
      int row = m0 + lr + r, col = 16 * j + lc;
      sv = (col <= row) ? sv : -1.0e30f;
      sacc[j][r] = sv;
      mx[r] = fmaxf(mx[r], sv);
    }
#pragma unroll
  for (int r = 0; r < 8; ++r) {
#pragma unroll
    for (int off = 8; off >= 1; off >>= 1)          // reduce within 16-lane half
      mx[r] = fmaxf(mx[r], __shfl_xor(mx[r], off, 32));
    sm[r] = 0.0f;
  }
#pragma unroll
  for (int j = 0; j < 8; ++j)
#pragma unroll
    for (int r = 0; r < 8; ++r) {
      float e = __expf(sacc[j][r] - mx[r]);
      sacc[j][r] = e;
      sm[r] += e;
    }
#pragma unroll
  for (int r = 0; r < 8; ++r) {
#pragma unroll
    for (int off = 8; off >= 1; off >>= 1)
      sm[r] += __shfl_xor(sm[r], off, 32);
    sm[r] = 1.0f / sm[r];
  }

  __syncthreads();   // all waves done reading sQ/sK before aliasing

  // Write P (bf16, A-layout row-major) and load V transposed (N-major B).
#pragma unroll
  for (int j = 0; j < 8; ++j)
#pragma unroll
    for (int r = 0; r < 8; ++r)
      sP[(size_t)(m0 + lr + r) * 136 + 16 * j + lc] = (__bf16)(sacc[j][r] * sm[r]);
#pragma unroll
  for (int i = 0; i < 32; ++i) {
    int idx = i * 256 + tid;
    int t = idx >> 6, d = idx & 63;
    sVt[d * 136 + t] = baseq[(size_t)t * C3 + 1536 + d];
  }
  __syncthreads();

  // ---- O = P V : 4 column tiles (D=64), 4 K-steps over T=128 ----
  v8f oacc[4];
#pragma unroll
  for (int j = 0; j < 4; ++j)
#pragma unroll
    for (int r = 0; r < 8; ++r) oacc[j][r] = 0.0f;

  for (int k0 = 0; k0 < 128; k0 += 32) {
    v16bf ap = lds_frag_a(sP, 136, m0, k0);
#pragma unroll
    for (int j = 0; j < 4; ++j) {
      v16bf bv = lds_frag_b(sVt, 136, 16 * j, k0);
      oacc[j] = __builtin_amdgcn_wmma_f32_16x16x32_bf16(
          false, ap, false, bv, (short)0, oacc[j], false, false);
    }
  }

  // Store O in (B,T,H*D) layout -> ready for the projection GEMM.
#pragma unroll
  for (int j = 0; j < 4; ++j)
#pragma unroll
    for (int r = 0; r < 8; ++r) {
      size_t row = (size_t)b * T + m0 + lr + r;
      attnout[row * 768 + h * 64 + 16 * j + lc] = bf_bits((__bf16)oacc[j][r]);
    }
}

// ---------------------------------------------------------------------------
extern "C" void kernel_launch(void* const* d_in, const int* in_sizes, int n_in,
                              void* d_out, int out_size, void* d_ws, size_t ws_size,
                              hipStream_t stream) {
  (void)in_sizes; (void)n_in; (void)out_size; (void)ws_size;
  const float* x      = (const float*)d_in[0];   // (128,128,768)
  const float* w_qkv  = (const float*)d_in[1];   // (768,2304)
  const float* w_proj = (const float*)d_in[2];   // (768,768)
  float* out = (float*)d_out;                    // (128,128,768) fp32

  const int B = 128, T = 128, C = 768;
  const int M = B * T;          // 16384
  const int N3 = 3 * C;         // 2304

  // Workspace: bf16 QKV (75.5 MB) + bf16 attention output (25 MB).
  unsigned short* qkv  = (unsigned short*)d_ws;
  unsigned short* attn = qkv + (size_t)M * N3;

  dim3 blk(256);
  // 1) QKV projection: (16384x768) @ (768x2304) -> bf16
  gemm_bf16_wmma<float, float, true>
      <<<dim3(M / 128, N3 / 128), blk, 0, stream>>>(x, w_qkv, (void*)qkv, N3, C);
  // 2) causal attention per (b,h)
  attn_wmma<<<dim3(B, 12), blk, 0, stream>>>(qkv, attn);
  // 3) output projection: (16384x768) @ (768x768) -> fp32
  gemm_bf16_wmma<unsigned short, float, false>
      <<<dim3(M / 128, C / 128), blk, 0, stream>>>(attn, w_proj, (void*)out, C, C);
}
